// AttnLayer_71382356460296
// MI455X (gfx1250) — matrix operand order repaired and checked by reading, not compiled, not run on hardware
//
#include <hip/hip_runtime.h>
#include <math.h>

typedef __attribute__((ext_vector_type(8)))  float  v8f;
typedef __attribute__((ext_vector_type(4)))  __bf16 v4bf;
typedef __attribute__((ext_vector_type(8)))  __bf16 v8bf;
typedef __attribute__((ext_vector_type(16))) __bf16 v16bf;

// bf16 WMMA: D = A(16x32) * B(32x16) + C, fp32 accumulate
#define WMMA_BF16(a, b, c) \
  __builtin_amdgcn_wmma_f32_16x16x32_bf16(false, (a), false, (b), (short)0, (c), false, false)

static constexpr int Hh  = 32;     // q heads
static constexpr int KVh = 8;      // kv heads (GQA rep = 4)
static constexpr int DHd = 64;     // head dim
static constexpr int WIN = 1024;   // attention window
static constexpr int Bb  = 2;
static constexpr int Tt  = 4096;
static constexpr int Dd  = 2048;
static constexpr int TCc = 4096;

// ---------------------------------------------------------------------------
// Async global -> LDS copy (CDNA5 GLOBAL_LOAD_ASYNC_TO_LDS_B128, ASYNCcnt)
// ---------------------------------------------------------------------------
__device__ __forceinline__ void async_load_b128(unsigned lds_addr,
                                                const void* sbase,
                                                unsigned voffset) {
  asm volatile("global_load_async_to_lds_b128 %0, %1, %2"
               :
               : "v"(lds_addr), "v"(voffset), "s"(sbase)
               : "memory");
}
__device__ __forceinline__ void s_wait_asynccnt_4() {
  asm volatile("s_wait_asynccnt 0x4" ::: "memory");
}
__device__ __forceinline__ void s_wait_asynccnt_0() {
  asm volatile("s_wait_asynccnt 0x0" ::: "memory");
}

// ---------------------------------------------------------------------------
// fp32 -> (hi, lo) bf16 decomposition (RNE both times): x ~= hi + lo
// ---------------------------------------------------------------------------
__device__ inline void split4(const float4& v, v4bf& h, v4bf& l) {
  float f[4] = {v.x, v.y, v.z, v.w};
#pragma unroll
  for (int i = 0; i < 4; ++i) {
    const __bf16 hb = (__bf16)f[i];
    h[i] = hb;
    l[i] = (__bf16)(f[i] - (float)hb);
  }
}

// A-frag (16x32, M x K), LDS row stride 32 bf16: lanes 0-15 hold K 0..7 &
// 16..23, lanes 16-31 hold K 8..15 & 24..31 (ISA 16-bit A layout).
__device__ inline v16bf load_fragA(const __bf16* S, int row, int half) {
  const v8bf p0 = *(const v8bf*)(S + row * 32 + half * 8);
  const v8bf p1 = *(const v8bf*)(S + row * 32 + 16 + half * 8);
  return __builtin_shufflevector(p0, p1, 0, 1, 2, 3, 4, 5, 6, 7,
                                 8, 9, 10, 11, 12, 13, 14, 15);
}

// B-frag (32x16, K x N) from an n-major slab with row stride 32:
// lane n; lanes 0-15 hold K 0..7 & 16..23, lanes 16-31 hold K 8..15 & 24..31.
__device__ inline v16bf load_fragB(const __bf16* S, int row, int half) {
  const v8bf p0 = *(const v8bf*)(S + row * 32 + half * 8);
  const v8bf p1 = *(const v8bf*)(S + row * 32 + 16 + half * 8);
  return __builtin_shufflevector(p0, p1, 0, 1, 2, 3, 4, 5, 6, 7,
                                 8, 9, 10, 11, 12, 13, 14, 15);
}

// Same as load_fragB but for a [s][d] slab with row stride 64, K-chunk dbase.
__device__ inline v16bf load_fragB64(const __bf16* S, int row, int dbase, int half) {
  const v8bf p0 = *(const v8bf*)(S + row * 64 + dbase + half * 8);
  const v8bf p1 = *(const v8bf*)(S + row * 64 + dbase + 16 + half * 8);
  return __builtin_shufflevector(p0, p1, 0, 1, 2, 3, 4, 5, 6, 7,
                                 8, 9, 10, 11, 12, 13, 14, 15);
}

// ---------------------------------------------------------------------------
// Split-precision GEMM: C[m,n] = sum_k A[m,k] * W[n,k]  (A @ W.T), fp32 I/O.
// bf16x3 via v_wmma_f32_16x16x32_bf16; raw fp32 slabs double-buffered in LDS
// and filled with GLOBAL_LOAD_ASYNC_TO_LDS_B128 one K-slab ahead.
// Block: 256 threads = 8 waves; tile 64(M) x 64(N); K-slab 32.
// ---------------------------------------------------------------------------
template <bool DO_ROPE>
__global__ __launch_bounds__(256) void gemm_bf16x3(const float* __restrict__ A,
                                                   const float* __restrict__ W,
                                                   float* __restrict__ Cout,
                                                   int K) {
  __shared__ float  rawA[2][64 * 32];
  __shared__ float  rawB[2][64 * 32];
  __shared__ __bf16 Ah[64 * 32];
  __shared__ __bf16 Al[64 * 32];
  __shared__ __bf16 Bh[64 * 32];
  __shared__ __bf16 Bl[64 * 32];

  const int tid     = threadIdx.x;
  const int lane    = tid & 31;
  const int wave    = tid >> 5;
  const int m0      = blockIdx.x * 64;
  const int n0      = blockIdx.y * 64;
  const int mt      = (wave >> 1) * 16;        // wave's M sub-tile
  const int ntA     = ((wave & 1) * 2) * 16;   // wave's first N sub-tile
  const int ntB     = ntA + 16;                // wave's second N sub-tile
  const int colLane = lane & 15;
  const int half    = lane >> 4;

  // Issue one 64x32 fp32 slab of A and W into raw[buf] (4 async ops / wave-lane)
  auto issue_slab = [&](int buf, int k0) {
#pragma unroll
    for (int i = 0; i < 2; ++i) {
      const int f   = tid + i * 256;           // 0..511 float4 slots
      const int row = f >> 3;
      const int c4  = f & 7;
      const unsigned voff = (unsigned)(((unsigned)row * (unsigned)K + c4 * 4) * 4u);
      async_load_b128((unsigned)(uintptr_t)&rawA[buf][f * 4],
                      A + (size_t)m0 * K + k0, voff);
      async_load_b128((unsigned)(uintptr_t)&rawB[buf][f * 4],
                      W + (size_t)n0 * K + k0, voff);
    }
  };

  v8f cA = {};
  v8f cB = {};

  issue_slab(0, 0);
  const int nIter = K / 32;
  for (int it = 0; it < nIter; ++it) {
    const int cur = it & 1;
    const int k0  = it * 32;
    if (it + 1 < nIter) {
      issue_slab(cur ^ 1, k0 + 32);   // prefetch next slab while cur completes
      s_wait_asynccnt_4();            // retire the 4 older (cur) copies
    } else {
      s_wait_asynccnt_0();
    }
    __syncthreads();                  // raw[cur] visible; prev frag reads done

    // Convert raw[cur] -> bf16 hi/lo slabs
#pragma unroll
    for (int i = 0; i < 2; ++i) {
      const int f = tid + i * 256;
      const float4 av = *(const float4*)&rawA[cur][f * 4];
      const float4 wv = *(const float4*)&rawB[cur][f * 4];
      v4bf ah, al, bh, bl;
      split4(av, ah, al);
      split4(wv, bh, bl);
      *(v4bf*)(Ah + f * 4) = ah;
      *(v4bf*)(Al + f * 4) = al;
      *(v4bf*)(Bh + f * 4) = bh;
      *(v4bf*)(Bl + f * 4) = bl;
    }
    __syncthreads();

    const v16bf aH  = load_fragA(Ah, mt + colLane, half);
    const v16bf aL  = load_fragA(Al, mt + colLane, half);
    const v16bf bH0 = load_fragB(Bh, ntA + colLane, half);
    const v16bf bL0 = load_fragB(Bl, ntA + colLane, half);
    const v16bf bH1 = load_fragB(Bh, ntB + colLane, half);
    const v16bf bL1 = load_fragB(Bl, ntB + colLane, half);

    cA = WMMA_BF16(aL, bH0, cA);   // low-order cross terms first
    cA = WMMA_BF16(aH, bL0, cA);
    cA = WMMA_BF16(aH, bH0, cA);
    cB = WMMA_BF16(aL, bH1, cB);
    cB = WMMA_BF16(aH, bL1, cB);
    cB = WMMA_BF16(aH, bH1, cB);
  }

  // Epilogue: C-frag layout = lane -> column N, VGPR r -> row r (+8 hi half)
#pragma unroll
  for (int r = 0; r < 8; ++r) {
    const int rowA = r + 8 * half;
    const int mg   = m0 + mt + rowA;     // global row in [0, B*T)
    const int b    = mg / Tt;
    const int t    = mg - b * Tt;        // RoPE position (pos_q == t)

    float vals[2];
    vals[0] = cA[r];
    vals[1] = cB[r];
    int nb[2];
    nb[0] = n0 + ntA;
    nb[1] = n0 + ntB;

#pragma unroll
    for (int f = 0; f < 2; ++f) {
      const int n = nb[f] + colLane;
      if (DO_ROPE) {
        const int d = n & 63;
        const int h = n >> 6;
        const float inv = __expf(-0.2878231366242557f * (float)(d >> 1));
        const float fr  = (float)t * inv;
        float sn, cs;
        __sincosf(fr, &sn, &cs);
        const float v     = vals[f];
        const float other = __shfl_xor(v, 1);
        const float o = (d & 1) ? (other * sn + v * cs)   // o2 = x1*sin + x2*cos
                                : (v * cs - other * sn);  // o1 = x1*cos - x2*sin
        Cout[(((size_t)b * Hh + h) * Tt + t) * DHd + d] = o;
      } else {
        Cout[(size_t)mg * Dd + n] = vals[f];
      }
    }
  }
}

// ---------------------------------------------------------------------------
// Flash-style windowed causal attention, bf16x3 WMMA throughout, fp32 softmax.
// Block: 128 threads = 4 waves = the 4 q-heads sharing one kv-head (GQA).
// s-step = 32 so the P@V K-dim matches the 16x16x32 bf16 WMMA shape.
// K staged [s][d] hi/lo; V staged transposed [d][s] hi/lo for contiguous
// B-fragments; P transposed through per-wave LDS as bf16 hi/lo.
// ---------------------------------------------------------------------------
__global__ __launch_bounds__(128) void attn_kernel(const float* __restrict__ qbuf,
                                                   const float* __restrict__ kc,
                                                   const float* __restrict__ vc,
                                                   float* __restrict__ ob) {
  __shared__ __bf16 Kh[32 * 64];
  __shared__ __bf16 Kl[32 * 64];
  __shared__ __bf16 Vh[64 * 32];   // transposed [d][s]
  __shared__ __bf16 Vl[64 * 32];
  __shared__ __bf16 Ph[4][16 * 32];
  __shared__ __bf16 Pl[4][16 * 32];

  const int tid     = threadIdx.x;
  const int lane    = tid & 31;
  const int wave    = tid >> 5;
  const int t0      = blockIdx.x * 16;
  const int g       = blockIdx.y;          // kv head
  const int b       = blockIdx.z;
  const int h       = g * 4 + wave;        // q head (rep = H/KV = 4)
  const int colLane = lane & 15;
  const int half    = lane >> 4;

  // Q A-fragments, bf16 hi/lo, two K-chunks of 32 over DH=64
  v16bf qh[2], ql[2];
  {
    const float* qrow = qbuf + (((size_t)b * Hh + h) * Tt + (t0 + colLane)) * DHd;
#pragma unroll
    for (int c = 0; c < 2; ++c) {
#pragma unroll
      for (int j = 0; j < 2; ++j) {        // j=0 -> K 0..7, j=1 -> K 16..23
        const float* p = qrow + c * 32 + j * 16 + half * 8;
#pragma unroll
        for (int e = 0; e < 8; ++e) {
          const float x  = p[e];
          const __bf16 hb = (__bf16)x;
          qh[c][j * 8 + e] = hb;
          ql[c][j * 8 + e] = (__bf16)(x - (float)hb);
        }
      }
    }
  }

  v8f o0 = {}, o1 = {}, o2 = {}, o3 = {};
  float mr[8], lr[8];
#pragma unroll
  for (int r = 0; r < 8; ++r) {
    mr[r] = -__builtin_inff();
    lr[r] = 0.0f;
  }

  const int nTiles = ((t0 + 15) >> 5) + 1;
  const int nst    = nTiles < (WIN / 32) ? nTiles : (WIN / 32);
  const float scale = 0.125f;  // 1/sqrt(64)
  const size_t kvbase =
      ((size_t)b * KVh + g) * TCc * DHd + (size_t)(TCc - WIN) * DHd;

  for (int st = 0; st < nst; ++st) {
    const int s0 = st * 32;
    __syncthreads();
    // Stage K (row-major) and V (transposed) as bf16 hi/lo: 32x64 fp32 each
#pragma unroll
    for (int i = 0; i < 4; ++i) {
      const int f   = tid + i * 128;       // 0..511 float4 slots
      const int row = f >> 4;              // s-local
      const int c4  = f & 15;              // d/4
      const float4 kv4 = *(const float4*)(kc + kvbase + (size_t)(s0 + row) * DHd + c4 * 4);
      const float4 vv4 = *(const float4*)(vc + kvbase + (size_t)(s0 + row) * DHd + c4 * 4);
      v4bf khh, kll, vhh, vll;
      split4(kv4, khh, kll);
      split4(vv4, vhh, vll);
      *(v4bf*)(Kh + row * 64 + c4 * 4) = khh;
      *(v4bf*)(Kl + row * 64 + c4 * 4) = kll;
#pragma unroll
      for (int e = 0; e < 4; ++e) {
        Vh[(c4 * 4 + e) * 32 + row] = vhh[e];
        Vl[(c4 * 4 + e) * 32 + row] = vll[e];
      }
    }
    __syncthreads();

    // Scores: two 16x16 tiles (s0..s0+15, s0+16..s0+31), bf16x3 over DH=64
    v8f S0 = {}, S1 = {};
#pragma unroll
    for (int c = 0; c < 2; ++c) {
      const v16bf kb0h = load_fragB64(Kh, colLane,      c * 32, half);
      const v16bf kb0l = load_fragB64(Kl, colLane,      c * 32, half);
      const v16bf kb1h = load_fragB64(Kh, colLane + 16, c * 32, half);
      const v16bf kb1l = load_fragB64(Kl, colLane + 16, c * 32, half);
      S0 = WMMA_BF16(ql[c], kb0h, S0);
      S0 = WMMA_BF16(qh[c], kb0l, S0);
      S0 = WMMA_BF16(qh[c], kb0h, S0);
      S1 = WMMA_BF16(ql[c], kb1h, S1);
      S1 = WMMA_BF16(qh[c], kb1l, S1);
      S1 = WMMA_BF16(qh[c], kb1h, S1);
    }

    // Online softmax over 32 columns (fp32, exact)
    float alpha[8];
#pragma unroll
    for (int r = 0; r < 8; ++r) {
      const int row = r + 8 * half;
      const int tq  = t0 + row;
      float v0 = S0[r] * scale;
      float v1 = S1[r] * scale;
      if (s0 + colLane > tq)      v0 = -__builtin_inff();
      if (s0 + 16 + colLane > tq) v1 = -__builtin_inff();

      float mx = fmaxf(v0, v1);
      mx = fmaxf(mx, __shfl_xor(mx, 1));
      mx = fmaxf(mx, __shfl_xor(mx, 2));
      mx = fmaxf(mx, __shfl_xor(mx, 4));
      mx = fmaxf(mx, __shfl_xor(mx, 8));
      const float mnew = fmaxf(mr[r], mx);
      const float a    = __expf(mr[r] - mnew);
      const float p0   = __expf(v0 - mnew);
      const float p1   = __expf(v1 - mnew);
      float ps = p0 + p1;
      ps += __shfl_xor(ps, 1);
      ps += __shfl_xor(ps, 2);
      ps += __shfl_xor(ps, 4);
      ps += __shfl_xor(ps, 8);
      lr[r]    = lr[r] * a + ps;
      mr[r]    = mnew;
      alpha[r] = a;

      // P transposed into A-frag layout via per-wave LDS, bf16 hi/lo split
      const __bf16 p0h = (__bf16)p0;
      const __bf16 p1h = (__bf16)p1;
      Ph[wave][row * 32 + colLane]      = p0h;
      Pl[wave][row * 32 + colLane]      = (__bf16)(p0 - (float)p0h);
      Ph[wave][row * 32 + 16 + colLane] = p1h;
      Pl[wave][row * 32 + 16 + colLane] = (__bf16)(p1 - (float)p1h);
    }

    // Rescale O accumulators
#pragma unroll
    for (int r = 0; r < 8; ++r) {
      o0[r] = o0[r] * alpha[r];
      o1[r] = o1[r] * alpha[r];
      o2[r] = o2[r] * alpha[r];
      o3[r] = o3[r] * alpha[r];
    }

    // O += P @ V : K-dim = 32 = one bf16 WMMA per term; 4 d-chunks x 3 terms
    const v16bf aPh = load_fragA(Ph[wave], colLane, half);
    const v16bf aPl = load_fragA(Pl[wave], colLane, half);
    {
      const v16bf vbh0 = load_fragB(Vh, 0 * 16 + colLane, half);
      const v16bf vbl0 = load_fragB(Vl, 0 * 16 + colLane, half);
      o0 = WMMA_BF16(aPl, vbh0, o0);
      o0 = WMMA_BF16(aPh, vbl0, o0);
      o0 = WMMA_BF16(aPh, vbh0, o0);
      const v16bf vbh1 = load_fragB(Vh, 1 * 16 + colLane, half);
      const v16bf vbl1 = load_fragB(Vl, 1 * 16 + colLane, half);
      o1 = WMMA_BF16(aPl, vbh1, o1);
      o1 = WMMA_BF16(aPh, vbl1, o1);
      o1 = WMMA_BF16(aPh, vbh1, o1);
      const v16bf vbh2 = load_fragB(Vh, 2 * 16 + colLane, half);
      const v16bf vbl2 = load_fragB(Vl, 2 * 16 + colLane, half);
      o2 = WMMA_BF16(aPl, vbh2, o2);
      o2 = WMMA_BF16(aPh, vbl2, o2);
      o2 = WMMA_BF16(aPh, vbh2, o2);
      const v16bf vbh3 = load_fragB(Vh, 3 * 16 + colLane, half);
      const v16bf vbl3 = load_fragB(Vl, 3 * 16 + colLane, half);
      o3 = WMMA_BF16(aPl, vbh3, o3);
      o3 = WMMA_BF16(aPh, vbl3, o3);
      o3 = WMMA_BF16(aPh, vbh3, o3);
    }
  }

  // Normalize and store directly in (B, T, H*DH) layout for the O-projection
#pragma unroll
  for (int r = 0; r < 8; ++r) {
    const int tq  = t0 + r + 8 * half;
    const float rl = 1.0f / lr[r];
    float* orow = ob + ((size_t)b * Tt + tq) * (Hh * DHd) + h * DHd;
    orow[0 + colLane]  = o0[r] * rl;
    orow[16 + colLane] = o1[r] * rl;
    orow[32 + colLane] = o2[r] * rl;
    orow[48 + colLane] = o3[r] * rl;
  }
}

// ---------------------------------------------------------------------------
extern "C" void kernel_launch(void* const* d_in, const int* in_sizes, int n_in,
                              void* d_out, int out_size, void* d_ws, size_t ws_size,
                              hipStream_t stream) {
  const float* x  = (const float*)d_in[0];
  const float* kc = (const float*)d_in[1];
  const float* vc = (const float*)d_in[2];
  const float* Wq = (const float*)d_in[3];
  const float* Wo = (const float*)d_in[4];
  float* out = (float*)d_out;

  float* qbuf = (float*)d_ws;                                // (B,H,T,DH) = 67 MB
  float* abuf = qbuf + (size_t)Bb * Hh * Tt * DHd;           // (B,T,H*DH) = 67 MB

  dim3 gGemm((Bb * Tt) / 64, Dd / 64);                       // (128, 32)
  gemm_bf16x3<true><<<gGemm, 256, 0, stream>>>(x, Wq, qbuf, Dd);

  dim3 gAttn(Tt / 16, KVh, Bb);                              // (256, 8, 2)
  attn_kernel<<<gAttn, 128, 0, stream>>>(qbuf, kc, vc, abuf);

  gemm_bf16x3<false><<<gGemm, 256, 0, stream>>>(abuf, Wo, out, Dd);
}
